// LigerGatedLinearAttention_66374424592748
// MI455X (gfx1250) — matrix operand order repaired and checked by reading, not compiled, not run on hardware
//
#include <hip/hip_runtime.h>
#include <hip/hip_bf16.h>
#include <math.h>

// ---------------------------------------------------------------------------
// Types
// ---------------------------------------------------------------------------
typedef __bf16 bf16_t;
typedef bf16_t v8bf  __attribute__((ext_vector_type(8)));
typedef bf16_t v16bf __attribute__((ext_vector_type(16)));
typedef float  v8f   __attribute__((ext_vector_type(8)));

#define SEQ    2048
#define HIDDEN 2048
#define NH     16
#define NKV    4
#define HD     128
#define QDIM   (NH * HD)   // 2048
#define KDIM   (NKV * HD)  // 512
#define GLA_C  64
#define GLA_NC (SEQ / GLA_C)

// ---------------------------------------------------------------------------
// gfx1250 async load-to-LDS (ASYNCcnt path), with portable fallback
// ---------------------------------------------------------------------------
#define AS1 __attribute__((address_space(1)))
#define AS3 __attribute__((address_space(3)))

typedef int v4i_b128 __attribute__((vector_size(16)));  // matches builtin V4i

#if __has_builtin(__builtin_amdgcn_global_load_async_to_lds_b128)
#define HAVE_ASYNC_LDS 1
#endif

__device__ inline void async_copy_b128(const bf16_t* g, bf16_t* l) {
#if defined(HAVE_ASYNC_LDS)
  __builtin_amdgcn_global_load_async_to_lds_b128(
      (AS1 v4i_b128*)(void*)g, (AS3 v4i_b128*)(void*)l, 0, 0);
#else
  *(v8bf*)l = *(const v8bf*)g;
#endif
}

#if __has_builtin(__builtin_amdgcn_s_wait_asynccnt)
#define WAIT_ASYNC(n) __builtin_amdgcn_s_wait_asynccnt(n)
#else
#define WAIT_ASYNC(n) asm volatile("s_wait_asynccnt %0" ::"i"(n) : "memory")
#endif

__device__ inline v8f v8f_zero() {
  v8f z;
#pragma unroll
  for (int i = 0; i < 8; ++i) z[i] = 0.f;
  return z;
}

__device__ inline bf16_t to_bf16(float x) { return (bf16_t)x; }

// ---------------------------------------------------------------------------
// WMMA helpers (gfx1250: wave32, 16x16x32 bf16 -> f32)
// ---------------------------------------------------------------------------
__device__ inline v8f wmma_bf16(v16bf a, v16bf b, v8f c) {
  return __builtin_amdgcn_wmma_f32_16x16x32_bf16(
      false, a, false, b, (short)0, c, false, false);
}

// A-matrix 16x32 bf16 fragment from row-major A (K contiguous).
__device__ inline v16bf load_a_frag(const bf16_t* __restrict__ A, int lda,
                                    int m, int k0, int lane) {
  int hi  = lane >> 4;
  int row = m + (lane & 15);
  const bf16_t* p = A + (size_t)row * lda + k0 + hi * 8;
  v8bf lo = *(const v8bf*)p;
  v8bf h2 = *(const v8bf*)(p + 16);
  v16bf f;
#pragma unroll
  for (int i = 0; i < 8; ++i) { f[i] = lo[i]; f[i + 8] = h2[i]; }
  return f;
}

// B-matrix 32x16 bf16 fragment from BT stored N-major (BT[n][k]).
__device__ inline v16bf load_b_frag(const bf16_t* __restrict__ BT, int ldb,
                                    int n, int k0, int lane) {
  int hi  = lane >> 4;
  int row = n + (lane & 15);
  const bf16_t* p = BT + (size_t)row * ldb + k0 + hi * 8;
  v8bf lo = *(const v8bf*)p;
  v8bf h2 = *(const v8bf*)(p + 16);
  v16bf f;
#pragma unroll
  for (int i = 0; i < 8; ++i) { f[i] = lo[i]; f[i + 8] = h2[i]; }
  return f;
}

__device__ inline void lds_fence_wave() {
  __builtin_amdgcn_wave_barrier();
  asm volatile("s_wait_dscnt 0" ::: "memory");
  __builtin_amdgcn_wave_barrier();
}

// ---------------------------------------------------------------------------
// Reductions (wave32)
// ---------------------------------------------------------------------------
__device__ inline float wave_reduce_max(float v) {
#pragma unroll
  for (int off = 16; off > 0; off >>= 1) v = fmaxf(v, __shfl_xor(v, off, 32));
  return v;
}
__device__ inline float wave_reduce_sum(float v) {
#pragma unroll
  for (int off = 16; off > 0; off >>= 1) v += __shfl_xor(v, off, 32);
  return v;
}
__device__ inline float half_reduce_max(float v) {
#pragma unroll
  for (int off = 8; off > 0; off >>= 1) v = fmaxf(v, __shfl_xor(v, off, 32));
  return v;
}
__device__ inline float half_reduce_sum(float v) {
#pragma unroll
  for (int off = 8; off > 0; off >>= 1) v += __shfl_xor(v, off, 32);
  return v;
}

// ---------------------------------------------------------------------------
// Conversion / transpose kernels
// ---------------------------------------------------------------------------
__global__ void f32_to_bf16_kernel(const float* __restrict__ in,
                                   bf16_t* __restrict__ out, int n) {
  int i = blockIdx.x * blockDim.x + threadIdx.x;
  if (i < n) out[i] = to_bf16(in[i]);
}

__global__ void transpose_bf16_kernel(const float* __restrict__ W,
                                      bf16_t* __restrict__ WT, int K, int N) {
  int i = blockIdx.x * blockDim.x + threadIdx.x;
  if (i >= K * N) return;
  int k = i / N, n = i - k * N;
  WT[(size_t)n * K + k] = to_bf16(W[i]);
}

__global__ void v_transpose_kernel(const float* __restrict__ V,
                                   bf16_t* __restrict__ VT) {
  int i = blockIdx.x * blockDim.x + threadIdx.x;
  if (i >= SEQ * KDIM) return;
  int pos = i / KDIM;
  int rem = i - pos * KDIM;
  int kvh = rem / HD;
  int d   = rem - kvh * HD;
  VT[((size_t)kvh * HD + d) * SEQ + pos] = to_bf16(V[i]);
}

// ---------------------------------------------------------------------------
// WMMA GEMM with double-buffered async LDS staging.
// C[M][N] (f32) = A[M][K] (bf16 row-major) x BT[N][K] (bf16).
// Block: 256 thr / 8 waves; tile 128(M) x 64(N); K-step 32.
// ---------------------------------------------------------------------------
#define GEMM_BM 128
#define GEMM_BN 64
#define GEMM_BK 32

__global__ __launch_bounds__(256) void gemm_bf16_wmma(
    const bf16_t* __restrict__ A, const bf16_t* __restrict__ BT,
    float* __restrict__ C, int M, int N, int K) {
  __shared__ bf16_t Abuf[2][GEMM_BM * GEMM_BK];
  __shared__ bf16_t Bbuf[2][GEMM_BN * GEMM_BK];

  int tid  = threadIdx.x;
  int lane = tid & 31;
  int wave = tid >> 5;
  int wr = wave & 3;   // M direction
  int wc = wave >> 2;  // N direction
  int bm = blockIdx.y * GEMM_BM;
  int bn = blockIdx.x * GEMM_BN;

  auto issue_copy = [&](int buf, int k0) {
#pragma unroll
    for (int c = 0; c < 3; ++c) {
      int chunk = tid + c * 256;  // 0..767
      if (chunk < 512) {          // A tile: 128 rows x 4 chunks
        int row = chunk >> 2, off = (chunk & 3) * 8;
        async_copy_b128(A + (size_t)(bm + row) * K + k0 + off,
                        &Abuf[buf][row * GEMM_BK + off]);
      } else {                    // B tile: 64 rows x 4 chunks
        int c2 = chunk - 512;
        int row = c2 >> 2, off = (c2 & 3) * 8;
        async_copy_b128(BT + (size_t)(bn + row) * K + k0 + off,
                        &Bbuf[buf][row * GEMM_BK + off]);
      }
    }
  };

  v8f acc00 = v8f_zero(), acc01 = v8f_zero();
  v8f acc10 = v8f_zero(), acc11 = v8f_zero();

  int nk = K / GEMM_BK;
  issue_copy(0, 0);
  for (int kt = 0; kt < nk; ++kt) {
    int buf = kt & 1;
    if (kt + 1 < nk) {
      issue_copy(buf ^ 1, (kt + 1) * GEMM_BK);
      WAIT_ASYNC(3);  // previous batch (3 async ops/wave) complete
    } else {
      WAIT_ASYNC(0);
    }
    __syncthreads();

    const bf16_t* At = Abuf[buf];
    const bf16_t* Bt = Bbuf[buf];
    v16bf a0 = load_a_frag(At, GEMM_BK, wr * 32, 0, lane);
    v16bf a1 = load_a_frag(At, GEMM_BK, wr * 32 + 16, 0, lane);
    v16bf b0 = load_b_frag(Bt, GEMM_BK, wc * 32, 0, lane);
    v16bf b1 = load_b_frag(Bt, GEMM_BK, wc * 32 + 16, 0, lane);
    acc00 = wmma_bf16(a0, b0, acc00);
    acc01 = wmma_bf16(a0, b1, acc01);
    acc10 = wmma_bf16(a1, b0, acc10);
    acc11 = wmma_bf16(a1, b1, acc11);
    __syncthreads();
  }

  int hi = lane >> 4, col = lane & 15;
  int m0 = bm + wr * 32, n0 = bn + wc * 32;
#pragma unroll
  for (int r = 0; r < 8; ++r) {
    int row0 = m0 + r + hi * 8;
    int row1 = m0 + 16 + r + hi * 8;
    C[(size_t)row0 * N + n0 + col]      = acc00[r];
    C[(size_t)row0 * N + n0 + 16 + col] = acc01[r];
    C[(size_t)row1 * N + n0 + col]      = acc10[r];
    C[(size_t)row1 * N + n0 + 16 + col] = acc11[r];
  }
}

// ---------------------------------------------------------------------------
// Post-projection: softmax over D + RoPE (NeoX rotate-half)
// ---------------------------------------------------------------------------
__global__ __launch_bounds__(128) void post_q_kernel(
    const float* __restrict__ Q, float* __restrict__ qs,
    bf16_t* __restrict__ sq) {
  __shared__ float xs[HD];
  __shared__ float red[4];
  int t = threadIdx.x;
  int lane = t & 31, wave = t >> 5;
  int h   = blockIdx.x & (NH - 1);
  int pos = blockIdx.x >> 4;
  size_t idx = (size_t)pos * QDIM + h * HD + t;
  float x = Q[idx];
  xs[t] = x;

  float m = wave_reduce_max(x);
  if (lane == 0) red[wave] = m;
  __syncthreads();
  m = fmaxf(fmaxf(red[0], red[1]), fmaxf(red[2], red[3]));
  float e = __expf(x - m);
  __syncthreads();
  float s = wave_reduce_sum(e);
  if (lane == 0) red[wave] = s;
  __syncthreads();
  s = red[0] + red[1] + red[2] + red[3];
  qs[idx] = e / s;

  int i = t & 63;
  float inv = __expf(-((float)(2 * i) / (float)HD) * 9.210340371976184f);
  float fr = (float)pos * inv;
  float c = __cosf(fr), sn = __sinf(fr);
  float part = xs[(t < 64) ? (t + 64) : (t - 64)];
  float ro = (t < 64) ? (x * c - part * sn) : (x * c + part * sn);
  sq[idx] = to_bf16(ro);
}

__global__ __launch_bounds__(128) void post_kv_kernel(
    const float* __restrict__ Kf, float* __restrict__ ks,
    float* __restrict__ gl, bf16_t* __restrict__ sk) {
  __shared__ float xs[HD];
  __shared__ float red[4];
  int t = threadIdx.x;
  int lane = t & 31, wave = t >> 5;
  int kvh = blockIdx.x & (NKV - 1);
  int pos = blockIdx.x >> 2;
  size_t idx = (size_t)pos * KDIM + kvh * HD + t;
  float x = Kf[idx];
  xs[t] = x;

  float m = wave_reduce_max(x);
  if (lane == 0) red[wave] = m;
  __syncthreads();
  m = fmaxf(fmaxf(red[0], red[1]), fmaxf(red[2], red[3]));
  float e = __expf(x - m);
  __syncthreads();
  float s = wave_reduce_sum(e);
  if (lane == 0) red[wave] = s;
  __syncthreads();
  s = red[0] + red[1] + red[2] + red[3];
  ks[idx] = e / s;

  float lsx = fminf(x, 0.f) - log1pf(__expf(-fabsf(x)));
  gl[idx] = lsx * 0.0625f;

  int i = t & 63;
  float inv = __expf(-((float)(2 * i) / (float)HD) * 9.210340371976184f);
  float fr = (float)pos * inv;
  float c = __cosf(fr), sn = __sinf(fr);
  float part = xs[(t < 64) ? (t + 64) : (t - 64)];
  float ro = (t < 64) ? (x * c - part * sn) : (x * c + part * sn);
  sk[idx] = to_bf16(ro);
}

// ---------------------------------------------------------------------------
// Flash-style causal attention with WMMA (one wave = 16 query rows).
// ---------------------------------------------------------------------------
__global__ __launch_bounds__(128) void flash_attn_kernel(
    const bf16_t* __restrict__ SQ, const bf16_t* __restrict__ SK,
    const bf16_t* __restrict__ VT, float* __restrict__ Y) {
  __shared__ bf16_t pbuf[4][16 * 32];
  const float scale = 0.08838834764831845f;  // 1/sqrt(128)

  int lane = threadIdx.x & 31;
  int wave = threadIdx.x >> 5;
  int hi = lane >> 4, col = lane & 15;
  int h = blockIdx.y;
  int kvh = h >> 2;
  int qt = blockIdx.x * 4 + wave;
  int qbase = qt * 16;

  const bf16_t* Qh = SQ + h * HD;
  const bf16_t* Kh = SK + kvh * HD;
  const bf16_t* Vh = VT + (size_t)kvh * HD * SEQ;
  bf16_t* pb = &pbuf[wave][0];

  v16bf aq[4];
#pragma unroll
  for (int dc = 0; dc < 4; ++dc) aq[dc] = load_a_frag(Qh, QDIM, qbase, dc * 32, lane);

  v8f acc[8];
#pragma unroll
  for (int j = 0; j < 8; ++j) acc[j] = v8f_zero();
  float mrow[8], lrow[8];
#pragma unroll
  for (int r = 0; r < 8; ++r) { mrow[r] = -1.0e30f; lrow[r] = 0.f; }

  int nkt = (qbase + 15) / 32 + 1;
  for (int kt = 0; kt < nkt; ++kt) {
    int k0 = kt * 32;
    v8f s0 = v8f_zero(), s1 = v8f_zero();
#pragma unroll
    for (int dc = 0; dc < 4; ++dc) {
      v16bf bk0 = load_b_frag(Kh, KDIM, k0, dc * 32, lane);
      v16bf bk1 = load_b_frag(Kh, KDIM, k0 + 16, dc * 32, lane);
      s0 = wmma_bf16(aq[dc], bk0, s0);
      s1 = wmma_bf16(aq[dc], bk1, s1);
    }
    float fac[8];
#pragma unroll
    for (int r = 0; r < 8; ++r) {
      int row = qbase + r + hi * 8;
      float e0 = (k0 + col <= row)      ? s0[r] * scale : -1.0e30f;
      float e1 = (k0 + 16 + col <= row) ? s1[r] * scale : -1.0e30f;
      float mt = half_reduce_max(fmaxf(e0, e1));
      float mnew = fmaxf(mrow[r], mt);
      float f = __expf(mrow[r] - mnew);
      float p0 = __expf(e0 - mnew);
      float p1 = __expf(e1 - mnew);
      float rs = half_reduce_sum(p0 + p1);
      lrow[r] = lrow[r] * f + rs;
      mrow[r] = mnew;
      fac[r] = f;
      pb[(r + hi * 8) * 32 + col]      = to_bf16(p0);
      pb[(r + hi * 8) * 32 + col + 16] = to_bf16(p1);
    }
#pragma unroll
    for (int j = 0; j < 8; ++j)
#pragma unroll
      for (int r = 0; r < 8; ++r) acc[j][r] *= fac[r];

    lds_fence_wave();
    v16bf ap = load_a_frag(pb, 32, 0, 0, lane);
#pragma unroll
    for (int j = 0; j < 8; ++j) {
      v16bf bv = load_b_frag(Vh, SEQ, j * 16, k0, lane);
      acc[j] = wmma_bf16(ap, bv, acc[j]);
    }
    lds_fence_wave();
  }

#pragma unroll
  for (int j = 0; j < 8; ++j)
#pragma unroll
    for (int r = 0; r < 8; ++r) {
      int row = qbase + r + hi * 8;
      Y[(size_t)row * QDIM + h * HD + j * 16 + col] = acc[j][r] / lrow[r];
    }
}

// ---------------------------------------------------------------------------
// Chunked GLA with WMMA. One workgroup per head, state SbT[dv][dk] (bf16)
// in dynamic LDS (~105 KB of 320 KB WGP budget).
// ---------------------------------------------------------------------------
extern __shared__ char gla_smem[];

__global__ __launch_bounds__(256) void gla_chunk_kernel(
    const float* __restrict__ qs, const float* __restrict__ ks,
    const float* __restrict__ vf, const float* __restrict__ gl,
    float* __restrict__ ogla) {
  bf16_t* SbT = (bf16_t*)gla_smem;
  bf16_t* qt  = SbT + HD * HD;
  bf16_t* ktl = qt + GLA_C * HD;
  bf16_t* keT = ktl + GLA_C * HD;
  bf16_t* vT  = keT + HD * GLA_C;
  bf16_t* Ab  = vT + HD * GLA_C;
  float*  egl = (float*)(Ab + GLA_C * GLA_C);

  int tid  = threadIdx.x;
  int lane = tid & 31;
  int wave = tid >> 5;
  int hi = lane >> 4, col = lane & 15;
  int h = blockIdx.x;
  int kvh = h >> 2;

  for (int i = tid; i < HD * HD; i += 256) SbT[i] = to_bf16(0.f);
  __syncthreads();

  for (int c = 0; c < GLA_NC; ++c) {
    int c0 = c * GLA_C;

    if (tid < HD) {
      int d = tid;
      float bl = 0.f;
      for (int t2 = 0; t2 < GLA_C; ++t2)
        bl += gl[(size_t)(c0 + t2) * KDIM + kvh * HD + d];
      egl[d] = __expf(bl);
      float b = 0.f;
      for (int t2 = 0; t2 < GLA_C; ++t2) {
        size_t kb = (size_t)(c0 + t2) * KDIM + kvh * HD + d;
        float g = gl[kb];
        b += g;
        float qv = qs[(size_t)(c0 + t2) * QDIM + h * HD + d];
        float kv = ks[kb];
        float vv = vf[kb];
        qt[t2 * HD + d]     = to_bf16(qv * __expf(b));
        ktl[t2 * HD + d]    = to_bf16(kv * __expf(-b));
        keT[d * GLA_C + t2] = to_bf16(kv * __expf(bl - b));
        vT[d * GLA_C + t2]  = to_bf16(vv);
      }
    }
    __syncthreads();

    {  // matmul1: A = tril(qt @ ktl^T)
      int mt = wave & 3;
      int nt0 = (wave >> 2) * 2;
      v8f a0 = v8f_zero(), a1 = v8f_zero();
#pragma unroll
      for (int dc = 0; dc < 4; ++dc) {
        v16bf af = load_a_frag(qt, HD, mt * 16, dc * 32, lane);
        v16bf b0 = load_b_frag(ktl, HD, nt0 * 16, dc * 32, lane);
        v16bf b1 = load_b_frag(ktl, HD, (nt0 + 1) * 16, dc * 32, lane);
        a0 = wmma_bf16(af, b0, a0);
        a1 = wmma_bf16(af, b1, a1);
      }
#pragma unroll
      for (int r = 0; r < 8; ++r) {
        int t2 = mt * 16 + r + hi * 8;
        int s0 = nt0 * 16 + col;
        int s1 = (nt0 + 1) * 16 + col;
        Ab[t2 * GLA_C + s0] = to_bf16((s0 <= t2) ? a0[r] : 0.f);
        Ab[t2 * GLA_C + s1] = to_bf16((s1 <= t2) ? a1[r] : 0.f);
      }
    }
    __syncthreads();

    {  // matmul2+3: o = Ab @ v + qt @ S
      int mt = wave & 3;
      int ntb = (wave >> 2) * 4;
      v8f oacc[4];
#pragma unroll
      for (int j = 0; j < 4; ++j) oacc[j] = v8f_zero();
#pragma unroll
      for (int sc = 0; sc < 2; ++sc) {
        v16bf af = load_a_frag(Ab, GLA_C, mt * 16, sc * 32, lane);
#pragma unroll
        for (int j = 0; j < 4; ++j) {
          v16bf bf = load_b_frag(vT, GLA_C, (ntb + j) * 16, sc * 32, lane);
          oacc[j] = wmma_bf16(af, bf, oacc[j]);
        }
      }
#pragma unroll
      for (int dc = 0; dc < 4; ++dc) {
        v16bf af = load_a_frag(qt, HD, mt * 16, dc * 32, lane);
#pragma unroll
        for (int j = 0; j < 4; ++j) {
          v16bf bf = load_b_frag(SbT, HD, (ntb + j) * 16, dc * 32, lane);
          oacc[j] = wmma_bf16(af, bf, oacc[j]);
        }
      }
#pragma unroll
      for (int j = 0; j < 4; ++j)
#pragma unroll
        for (int r = 0; r < 8; ++r) {
          int t2 = mt * 16 + r + hi * 8;
          int dv = (ntb + j) * 16 + col;
          ogla[(size_t)(c0 + t2) * QDIM + h * HD + dv] = oacc[j][r];
        }
    }

    v8f dacc[8];  // matmul4: dS = ke^T @ v
    {
#pragma unroll
      for (int j = 0; j < 8; ++j) dacc[j] = v8f_zero();
#pragma unroll
      for (int sc = 0; sc < 2; ++sc) {
        v16bf af = load_a_frag(keT, GLA_C, wave * 16, sc * 32, lane);
#pragma unroll
        for (int j = 0; j < 8; ++j) {
          v16bf bf = load_b_frag(vT, GLA_C, j * 16, sc * 32, lane);
          dacc[j] = wmma_bf16(af, bf, dacc[j]);
        }
      }
    }
    __syncthreads();

#pragma unroll
    for (int j = 0; j < 8; ++j)
#pragma unroll
      for (int r = 0; r < 8; ++r) {
        int dk = wave * 16 + r + hi * 8;
        int dv = j * 16 + col;
        float olds = (float)SbT[dv * HD + dk];
        SbT[dv * HD + dk] = to_bf16(olds * egl[dk] + dacc[j][r]);
      }
    __syncthreads();
  }
}

// ---------------------------------------------------------------------------
// Combine: o = 0.5*y + 0.5*o_gla -> bf16
// ---------------------------------------------------------------------------
__global__ void combine_kernel(const float* __restrict__ Y,
                               const float* __restrict__ OG,
                               bf16_t* __restrict__ OB, int n) {
  int i = blockIdx.x * blockDim.x + threadIdx.x;
  if (i < n) OB[i] = to_bf16(0.5f * Y[i] + 0.5f * OG[i]);
}

// ---------------------------------------------------------------------------
// Host launcher
// ---------------------------------------------------------------------------
extern "C" void kernel_launch(void* const* d_in, const int* in_sizes, int n_in,
                              void* d_out, int out_size, void* d_ws, size_t ws_size,
                              hipStream_t stream) {
  const float* hs = (const float*)d_in[0];
  const float* Wq = (const float*)d_in[1];
  const float* Wk = (const float*)d_in[2];
  const float* Wv = (const float*)d_in[3];
  const float* Wo = (const float*)d_in[4];
  float* out = (float*)d_out;

  char* ws = (char*)d_ws;
  size_t off = 0;
  auto take = [&](size_t bytes) -> char* {
    char* p = ws + off;
    off += (bytes + 255) & ~(size_t)255;
    return p;
  };
  const size_t NQ = (size_t)SEQ * QDIM;
  const size_t NK = (size_t)SEQ * KDIM;
  bf16_t* hsb = (bf16_t*)take(NQ * 2);
  bf16_t* wqT = (bf16_t*)take((size_t)HIDDEN * QDIM * 2);
  bf16_t* wkT = (bf16_t*)take((size_t)HIDDEN * KDIM * 2);
  bf16_t* wvT = (bf16_t*)take((size_t)HIDDEN * KDIM * 2);
  bf16_t* woT = (bf16_t*)take((size_t)QDIM * HIDDEN * 2);
  float*  Qf  = (float*)take(NQ * 4);
  float*  Kf  = (float*)take(NK * 4);
  float*  Vf  = (float*)take(NK * 4);
  float*  qsA = (float*)take(NQ * 4);
  float*  ksA = (float*)take(NK * 4);
  float*  glA = (float*)take(NK * 4);
  bf16_t* sqb = (bf16_t*)take(NQ * 2);
  bf16_t* skb = (bf16_t*)take(NK * 2);
  bf16_t* vtb = (bf16_t*)take(NK * 2);
  float*  yf  = (float*)take(NQ * 4);
  float*  og  = (float*)take(NQ * 4);
  bf16_t* ob  = (bf16_t*)take(NQ * 2);

  f32_to_bf16_kernel<<<(int)((NQ + 255) / 256), 256, 0, stream>>>(hs, hsb, (int)NQ);
  transpose_bf16_kernel<<<(int)((NQ + 255) / 256), 256, 0, stream>>>(Wq, wqT, HIDDEN, QDIM);
  transpose_bf16_kernel<<<(int)((NK + 255) / 256), 256, 0, stream>>>(Wk, wkT, HIDDEN, KDIM);
  transpose_bf16_kernel<<<(int)((NK + 255) / 256), 256, 0, stream>>>(Wv, wvT, HIDDEN, KDIM);
  transpose_bf16_kernel<<<(int)((NQ + 255) / 256), 256, 0, stream>>>(Wo, woT, QDIM, HIDDEN);

  {
    dim3 g(QDIM / 64, SEQ / 128);
    gemm_bf16_wmma<<<g, 256, 0, stream>>>(hsb, wqT, Qf, SEQ, QDIM, HIDDEN);
  }
  {
    dim3 g(KDIM / 64, SEQ / 128);
    gemm_bf16_wmma<<<g, 256, 0, stream>>>(hsb, wkT, Kf, SEQ, KDIM, HIDDEN);
    gemm_bf16_wmma<<<g, 256, 0, stream>>>(hsb, wvT, Vf, SEQ, KDIM, HIDDEN);
  }

  post_q_kernel<<<SEQ * NH, 128, 0, stream>>>(Qf, qsA, sqb);
  post_kv_kernel<<<SEQ * NKV, 128, 0, stream>>>(Kf, ksA, glA, skb);
  v_transpose_kernel<<<(int)((NK + 255) / 256), 256, 0, stream>>>(Vf, vtb);

  {
    dim3 g(SEQ / 64, NH);
    flash_attn_kernel<<<g, 128, 0, stream>>>(sqb, skb, vtb, yf);
  }

  {
    size_t lds = (size_t)(HD * HD + 2 * GLA_C * HD + 2 * HD * GLA_C +
                          GLA_C * GLA_C) * sizeof(bf16_t) + HD * sizeof(float);
    gla_chunk_kernel<<<NH, 256, lds, stream>>>(qsA, ksA, Vf, glA, og);
  }

  combine_kernel<<<(int)((NQ + 255) / 256), 256, 0, stream>>>(yf, og, ob, (int)NQ);
  {
    dim3 g(HIDDEN / 64, SEQ / 128);
    gemm_bf16_wmma<<<g, 256, 0, stream>>>(ob, woT, out, SEQ, HIDDEN, QDIM);
  }
  (void)in_sizes; (void)n_in; (void)out_size; (void)ws_size;
}
